// EdgeModel_9440338117437
// MI455X (gfx1250) — matrix operand order repaired and checked by reading, not compiled, not run on hardware
//
#include <hip/hip_runtime.h>
#include <hip/hip_bf16.h>

// CDNA5 / gfx1250 fused EdgeModel kernel, round 2.
// bf16 WMMA (v_wmma_f32_16x16x32_bf16) for both GEMMs, fp32 accumulate.
// Weights pre-transposed/converted to bf16 once into d_ws; B fragments are
// loaded straight from global (L2-resident) with no LDS staging.

typedef __attribute__((ext_vector_type(16))) __bf16 v16bf;
typedef __attribute__((ext_vector_type(8)))  float  v8f;

#define M_TILE   64
#define ASTRIDE  40    // 32 K + 8 pad (elems) -> 80B rows, 16B aligned
#define HSTRIDE  136   // 128 + 8 pad -> 272B rows, 16B aligned
#define KPAD1    544   // 17 * 32 padded in_dim

union FragU { v16bf bf; uint4 u[2]; };
union Pack8 { uint4 u; __bf16 h[8]; };

// ---------------- SE(3) helpers (fp32 VALU) ----------------
__device__ __forceinline__ void qmul(const float* a, const float* b, float* o) {
    o[0] = a[0]*b[0] - a[1]*b[1] - a[2]*b[2] - a[3]*b[3];
    o[1] = a[0]*b[1] + a[1]*b[0] + a[2]*b[3] - a[3]*b[2];
    o[2] = a[0]*b[2] - a[1]*b[3] + a[2]*b[0] + a[3]*b[1];
    o[3] = a[0]*b[3] + a[1]*b[2] - a[2]*b[1] + a[3]*b[0];
}
__device__ __forceinline__ void qrot(const float* q, const float* v, float* o) {
    float w = q[0], qx = q[1], qy = q[2], qz = q[3];
    float tx = 2.0f * (qy*v[2] - qz*v[1]);
    float ty = 2.0f * (qz*v[0] - qx*v[2]);
    float tz = 2.0f * (qx*v[1] - qy*v[0]);
    o[0] = v[0] + w*tx + (qy*tz - qz*ty);
    o[1] = v[1] + w*ty + (qz*tx - qx*tz);
    o[2] = v[2] + w*tz + (qx*ty - qy*tx);
}
__device__ __forceinline__ void se3_inv(const float* T, float* o) {
    float qi[4] = {T[0], -T[1], -T[2], -T[3]};
    float t[3];
    qrot(qi, T + 4, t);
    o[0] = qi[0]; o[1] = qi[1]; o[2] = qi[2]; o[3] = qi[3];
    o[4] = -t[0]; o[5] = -t[1]; o[6] = -t[2];
}
__device__ __forceinline__ void se3_mul(const float* A, const float* Bv, float* o) {
    qmul(A, Bv, o);
    float r[3];
    qrot(A, Bv + 4, r);
    o[4] = r[0] + A[4]; o[5] = r[1] + A[5]; o[6] = r[2] + A[6];
}

// ---------------- WMMA fragment loads ----------------
// 16-bit A-matrix 16x32 (ISA 7.12.2): lane<16 -> row=lane, K={0..7,16..23};
// lane>=16 -> row=lane-16, K={8..15,24..31}. LDS tile is row-major [row][K].
__device__ __forceinline__ v16bf load_frag_a(const __bf16* base, int row0, int stride) {
    const int lane = threadIdx.x & 31;
    const int row  = row0 + (lane & 15);
    const int kofs = (lane >> 4) << 3;           // 0 or 8
    const __bf16* p = base + row * stride + kofs;
    FragU f;
    f.u[0] = *(const uint4*)(p);                 // K = kofs .. kofs+7
    f.u[1] = *(const uint4*)(p + 16);            // K = kofs+16 .. kofs+23
    return f.bf;
}
// 16-bit B-matrix 32x16: lane<16 -> col=lane, K={0..7,16..23}; lane>=16 -> col=lane-16,
// K={8..15,24..31}. Weights are pre-transposed col-major [n][Kpad] bf16 in GLOBAL memory:
// each lane reads two 16B chunks 32B apart (two global_load_b128, L2-resident).
__device__ __forceinline__ v16bf load_frag_b_g(const __bf16* __restrict__ base,
                                               int col0, int kstep, int ldk) {
    const int lane = threadIdx.x & 31;
    const int col  = col0 + (lane & 15);
    const int kofs = (lane >> 4) << 3;           // 0 or 8
    const __bf16* p = base + (size_t)col * ldk + kstep + kofs;
    FragU f;
    f.u[0] = *(const uint4*)(p);
    f.u[1] = *(const uint4*)(p + 16);
    return f.bf;
}

// ---------------- weight prep: fp32 -> bf16, transposed, K padded ----------------
// W1t[n][kp], kp in [0,544): step=kp/32 maps to original W1 rows per segment
// (0..383 contiguous, 384..390 T_err (pad->0), 391..518 u). W2t[n][k] col-major.
__global__ __launch_bounds__(256)
void prep_weights(const float* __restrict__ W1, const float* __restrict__ W2,
                  __bf16* __restrict__ W1t, __bf16* __restrict__ W2t) {
    int i = blockIdx.x * 256 + threadIdx.x;
    if (i < 128 * KPAD1) {
        int n  = i / KPAD1;
        int kp = i % KPAD1;
        int step = kp >> 5, kk = kp & 31;
        float v = 0.0f;
        if (step < 12)        v = W1[(size_t)(step * 32 + kk) * 128 + n];
        else if (step == 12) { if (kk < 7) v = W1[(size_t)(384 + kk) * 128 + n]; }
        else                  v = W1[(size_t)(391 + (step - 13) * 32 + kk) * 128 + n];
        W1t[(size_t)n * KPAD1 + kp] = (__bf16)v;
    } else {
        int j = i - 128 * KPAD1;
        if (j < 128 * 128) {
            int n = j >> 7, k = j & 127;
            W2t[(size_t)n * 128 + k] = (__bf16)W2[(size_t)k * 128 + n];
        }
    }
}

// ---------------- fused kernel ----------------
__global__ __launch_bounds__(256)
void edge_model_fused(const float* __restrict__ xfeati,
                      const float* __restrict__ xfeatj,
                      const float* __restrict__ edge_feat,
                      const float* __restrict__ Ti,
                      const float* __restrict__ Tj,
                      const float* __restrict__ That,
                      const float* __restrict__ u,
                      const int*   __restrict__ batch,
                      const __bf16* __restrict__ W1t,
                      const float* __restrict__ b1,
                      const __bf16* __restrict__ W2t,
                      const float* __restrict__ b2,
                      float* __restrict__ out_edge,
                      float* __restrict__ out_that,
                      int E)
{
    __shared__ __align__(16) __bf16 lds_a   [2 * M_TILE * ASTRIDE];  // double buffer
    __shared__ __align__(16) __bf16 lds_terr[M_TILE * ASTRIDE];
    __shared__ __align__(16) __bf16 lds_h   [M_TILE * HSTRIDE];

    const int tid  = threadIdx.x;
    const int lane = tid & 31;
    const int wv   = tid >> 5;                   // wave 0..7, owns 16 output cols
    const int row0 = blockIdx.x * M_TILE;
    const int col0 = wv * 16;

    // ---- zero T_err staging tile (zero-padded K 7..31) ----
    {
        uint* z = (uint*)lds_terr;
        #pragma unroll
        for (int i = tid; i < M_TILE * ASTRIDE / 2; i += 256) z[i] = 0u;
    }
    __syncthreads();

    // ---- compute T_err = That * inv(Tj * inv(Ti)) per row ----
    if (tid < M_TILE) {
        int r = row0 + tid;
        if (r < E) {
            float ti[7], tj[7], th[7], tii[7], tij[7], tji[7], terr[7];
            #pragma unroll
            for (int i = 0; i < 7; ++i) {
                ti[i] = Ti[(size_t)r * 7 + i];
                tj[i] = Tj[(size_t)r * 7 + i];
                th[i] = That[(size_t)r * 7 + i];
            }
            se3_inv(ti, tii);
            se3_mul(tj, tii, tij);
            se3_inv(tij, tji);
            se3_mul(th, tji, terr);
            #pragma unroll
            for (int i = 0; i < 7; ++i)
                lds_terr[tid * ASTRIDE + i] = (__bf16)terr[i];
        }
    }

    v8f acc[4];
    const v8f vzero = {0.f, 0.f, 0.f, 0.f, 0.f, 0.f, 0.f, 0.f};
    #pragma unroll
    for (int m = 0; m < 4; ++m) acc[m] = vzero;

    // A-staging addressing (fixed per thread): 4 threads per row, 8 cols each
    const int arow = tid >> 2;
    const int ac   = (tid & 3) * 8;
    const int grow = row0 + arow;

    // =========== GEMM1: h = relu(feat_cat @ W1 + b1), 17 K-steps of 32 ===========
    // One barrier per step: stage buf[s&1]; sync; compute. The sync at step s-1
    // guarantees compute of step s-2 finished before buf[s&1] is overwritten.
    for (int step = 0; step < 17; ++step) {
        __bf16* abuf = lds_a + (step & 1) * (M_TILE * ASTRIDE);

        if (step != 12) {
            const float* asrc;
            int acol0;
            if (step < 12) {
                int seg = step >> 2;
                const float* s0 = (seg == 0) ? xfeati : (seg == 1) ? xfeatj : edge_feat;
                acol0 = (step & 3) * 32;
                asrc  = s0 + (size_t)grow * 128;
            } else {                              // u gather segment
                acol0 = (step - 13) * 32;
                asrc  = (grow < E) ? (u + (size_t)batch[grow] * 128) : u;
            }
            Pack8 pk;
            if (grow < E) {
                float4 f0 = *(const float4*)(asrc + acol0 + ac);
                float4 f1 = *(const float4*)(asrc + acol0 + ac + 4);
                pk.h[0] = (__bf16)f0.x; pk.h[1] = (__bf16)f0.y;
                pk.h[2] = (__bf16)f0.z; pk.h[3] = (__bf16)f0.w;
                pk.h[4] = (__bf16)f1.x; pk.h[5] = (__bf16)f1.y;
                pk.h[6] = (__bf16)f1.z; pk.h[7] = (__bf16)f1.w;
            } else {
                pk.u = make_uint4(0u, 0u, 0u, 0u);
            }
            *(uint4*)(abuf + arow * ASTRIDE + ac) = pk.u;
        }

        __syncthreads();

        const __bf16* abase = (step == 12) ? lds_terr : abuf;
        v16bf bf = load_frag_b_g(W1t, col0, step * 32, KPAD1);
        #pragma unroll
        for (int m = 0; m < 4; ++m) {
            v16bf af = load_frag_a(abase, m * 16, ASTRIDE);
            acc[m] = __builtin_amdgcn_wmma_f32_16x16x32_bf16(
                         false, af, false, bf, (short)0, acc[m], false, false);
        }
    }

    // ---- h = relu(acc + b1) -> LDS (bf16, row-major, A-layout for GEMM2) ----
    const int   crow  = (lane >> 4) * 8;         // +0 / +8 row split of C tile
    const int   ccol  = col0 + (lane & 15);
    const float bias1 = b1[ccol];
    #pragma unroll
    for (int m = 0; m < 4; ++m) {
        #pragma unroll
        for (int v = 0; v < 8; ++v) {
            float hv = fmaxf(acc[m][v] + bias1, 0.0f);
            int r = m * 16 + crow + v;
            lds_h[r * HSTRIDE + ccol] = (__bf16)hv;
        }
        acc[m] = vzero;
    }
    __syncthreads();                              // lds_h visible to all waves

    // ====== GEMM2: out = edge_feat + h @ W2 + b2 (barrier-free, B from L2) ======
    #pragma unroll
    for (int kb = 0; kb < 4; ++kb) {
        v16bf bf = load_frag_b_g(W2t, col0, kb * 32, 128);
        #pragma unroll
        for (int m = 0; m < 4; ++m) {
            v16bf af = load_frag_a(lds_h + kb * 32, m * 16, HSTRIDE);
            acc[m] = __builtin_amdgcn_wmma_f32_16x16x32_bf16(
                         false, af, false, bf, (short)0, acc[m], false, false);
        }
    }

    // ---- epilogue: residual + bias, store ----
    const float bias2 = b2[ccol];
    #pragma unroll
    for (int m = 0; m < 4; ++m) {
        #pragma unroll
        for (int v = 0; v < 8; ++v) {
            int r  = m * 16 + crow + v;
            int gr = row0 + r;
            if (gr < E) {
                size_t idx = (size_t)gr * 128 + ccol;
                out_edge[idx] = acc[m][v] + bias2 + edge_feat[idx];
            }
        }
    }

    // ---- second output: Tij_hat passthrough ----
    for (int e = tid; e < M_TILE * 7; e += 256) {
        int r = row0 + e / 7;
        if (r < E) {
            size_t idx = (size_t)r * 7 + (e % 7);
            out_that[idx] = That[idx];
        }
    }
}

extern "C" void kernel_launch(void* const* d_in, const int* in_sizes, int n_in,
                              void* d_out, int out_size, void* d_ws, size_t ws_size,
                              hipStream_t stream) {
    const float* xfeati    = (const float*)d_in[0];
    const float* xfeatj    = (const float*)d_in[1];
    const float* edge_feat = (const float*)d_in[2];
    const float* Ti        = (const float*)d_in[3];
    const float* Tj        = (const float*)d_in[4];
    const float* That      = (const float*)d_in[5];
    const float* u         = (const float*)d_in[6];
    const int*   batch     = (const int*)d_in[7];
    const float* W1        = (const float*)d_in[8];
    const float* b1        = (const float*)d_in[9];
    const float* W2        = (const float*)d_in[10];
    const float* b2        = (const float*)d_in[11];

    const int E = in_sizes[0] / 128;
    float* out_edge = (float*)d_out;
    float* out_that = out_edge + (size_t)E * 128;

    // workspace layout: W1t bf16 [128][544], then W2t bf16 [128][128]
    __bf16* W1t = (__bf16*)d_ws;
    __bf16* W2t = W1t + (size_t)128 * KPAD1;

    const int prep_elems = 128 * KPAD1 + 128 * 128;
    prep_weights<<<(prep_elems + 255) / 256, 256, 0, stream>>>(W1, W2, W1t, W2t);

    const int blocks = (E + M_TILE - 1) / M_TILE;
    edge_model_fused<<<blocks, 256, 0, stream>>>(
        xfeati, xfeatj, edge_feat, Ti, Tj, That, u, batch,
        W1t, b1, W2t, b2, out_edge, out_that, E);
}
